// HostGCN_31714038513704
// MI455X (gfx1250) — matrix operand order repaired and checked by reading, not compiled, not run on hardware
//
#include <hip/hip_runtime.h>
#include <hip/hip_bf16.h>
#include <math.h>

// ---------------------------------------------------------------------------
// GCN: 3x (GEMM + symmetric-norm aggregation) + BN/ReLU + log_softmax
// N=50000 nodes, E=800000 edges, widths 128 -> 128 -> 128 -> 16
// GEMMs: v_wmma_f32_16x16x32_f16 (wave32). W pre-swizzled to per-lane
// fragment order (contiguous 32B v16h global loads), A tile in LDS with
// b128 fragment loads, 64 rows/block -> 16 WMMA per wave, uniform-branch
// unguarded write-out fast path. All elementwise/aggregation kernels are
// float4-vectorized; aggregation = wave-per-edge + f32 atomics (L2 resident).
// ---------------------------------------------------------------------------

typedef __attribute__((ext_vector_type(16))) _Float16 v16h;
typedef __attribute__((ext_vector_type(8)))  _Float16 v8h;
typedef __attribute__((ext_vector_type(4)))  _Float16 v4h;
typedef __attribute__((ext_vector_type(8)))  float    v8f;

#define BN_EPS 1e-5f
#define MROWS  64            // rows per block (four 16-row WMMA subtiles)

// ---------------- degree / normalization ----------------

__global__ void k_fill_deg(float* __restrict__ deg, int n) {
  int i = blockIdx.x * blockDim.x + threadIdx.x;
  if (i < n) deg[i] = 1.0f;  // self-loop contributes 1 to every degree
}

__global__ void k_deg(const long long* __restrict__ dst, float* __restrict__ deg, int nE) {
  int i = blockIdx.x * blockDim.x + threadIdx.x;
  if (i < nE) atomicAdd(&deg[(size_t)dst[i]], 1.0f);
}

__global__ void k_dinv(const float* __restrict__ deg, float* __restrict__ dinv, int n) {
  int i = blockIdx.x * blockDim.x + threadIdx.x;
  if (i < n) dinv[i] = rsqrtf(fmaxf(deg[i], 1.0f));
}

// Swizzle W[128 x Hc] (f32) into WMMA B-fragment order (f16):
//   Wsw[((ct*4 + kc)*32 + lane)*16 + i] = W[k][ct*16 + (lane&15)],
//   k = kc*32 + (i&7) + 8*(lane>>4) + 16*(i>>3)  -> each wave's B operand is one 32B v16h.
__global__ void k_swizzle_w(const float* __restrict__ W, _Float16* __restrict__ Wsw, int Hc) {
  int o = blockIdx.x * blockDim.x + threadIdx.x;
  if (o >= 128 * Hc) return;
  int i    = o & 15;
  int lane = (o >> 4) & 31;
  int kc   = (o >> 9) & 3;
  int ct   = o >> 11;
  int k    = kc * 32 + (i & 7) + 8 * (lane >> 4) + 16 * (i >> 3);
  int col  = ct * 16 + (lane & 15);
  Wsw[o] = (_Float16)W[k * Hc + col];
}

// ---------------- elementwise (float4-vectorized, HC compile-time) ----------

// agg[i][f] = bias[f] + dinv[i]^2 * h[i][f]   (self-loop message + bias)
template <int HC>
__global__ void k_selfinit(const float* __restrict__ h, const float* __restrict__ dinv,
                           const float* __restrict__ bias, float* __restrict__ agg, int n) {
  int i4 = blockIdx.x * blockDim.x + threadIdx.x;          // float4 index
  if (i4 >= n * (HC / 4)) return;
  int r  = i4 / (HC / 4);
  int fq = i4 - r * (HC / 4);
  float di = dinv[r];
  float dd = di * di;
  float4 hv = ((const float4*)h)[i4];
  float4 bv = ((const float4*)bias)[fq];
  float4 o  = make_float4(bv.x + dd * hv.x, bv.y + dd * hv.y,
                          bv.z + dd * hv.z, bv.w + dd * hv.w);
  ((float4*)agg)[i4] = o;
}

// agg[dst] += (dinv[src]*dinv[dst]) * h[src]; wave per edge, float4 row reads
template <int HC>
__global__ void k_scatter(const float* __restrict__ h, const long long* __restrict__ src,
                          const long long* __restrict__ dst, const float* __restrict__ dinv,
                          float* __restrict__ agg, int nE) {
  int gid  = blockIdx.x * blockDim.x + threadIdx.x;
  int wid  = gid >> 5;
  int lane = threadIdx.x & 31;
  if (wid >= nE) return;
  long long s = src[wid];
  long long d = dst[wid];
  float w = dinv[(size_t)s] * dinv[(size_t)d];
  const float* hs = h + (size_t)s * HC;
  float*       ad = agg + (size_t)d * HC;
#pragma unroll
  for (int f = lane * 4; f < HC; f += 128) {
    float4 hv = *(const float4*)(hs + f);
    atomicAdd(&ad[f + 0], w * hv.x);
    atomicAdd(&ad[f + 1], w * hv.y);
    atomicAdd(&ad[f + 2], w * hv.z);
    atomicAdd(&ad[f + 3], w * hv.w);
  }
}

template <int HC>
__global__ void k_bn_relu(const float* __restrict__ x, const float* __restrict__ g,
                          const float* __restrict__ beta, const float* __restrict__ m,
                          const float* __restrict__ v, float* __restrict__ out, int n) {
  int i4 = blockIdx.x * blockDim.x + threadIdx.x;
  if (i4 >= n * (HC / 4)) return;
  int r  = i4 / (HC / 4);
  int fq = i4 - r * (HC / 4);
  float4 xv = ((const float4*)x)[i4];
  float4 gv = ((const float4*)g)[fq];
  float4 bv = ((const float4*)beta)[fq];
  float4 mv = ((const float4*)m)[fq];
  float4 vv = ((const float4*)v)[fq];
  float4 o;
  o.x = fmaxf((xv.x - mv.x) * rsqrtf(vv.x + BN_EPS) * gv.x + bv.x, 0.f);
  o.y = fmaxf((xv.y - mv.y) * rsqrtf(vv.y + BN_EPS) * gv.y + bv.y, 0.f);
  o.z = fmaxf((xv.z - mv.z) * rsqrtf(vv.z + BN_EPS) * gv.z + bv.z, 0.f);
  o.w = fmaxf((xv.w - mv.w) * rsqrtf(vv.w + BN_EPS) * gv.w + bv.w, 0.f);
  ((float4*)out)[i4] = o;
}

__global__ void k_logsoftmax16(const float* __restrict__ x, float* __restrict__ out, int n) {
  int r = blockIdx.x * blockDim.x + threadIdx.x;
  if (r >= n) return;
  const float4* xr = (const float4*)(x + (size_t)r * 16);
  float4 q0 = xr[0], q1 = xr[1], q2 = xr[2], q3 = xr[3];
  float v[16] = {q0.x, q0.y, q0.z, q0.w, q1.x, q1.y, q1.z, q1.w,
                 q2.x, q2.y, q2.z, q2.w, q3.x, q3.y, q3.z, q3.w};
  float mx = v[0];
#pragma unroll
  for (int j = 1; j < 16; ++j) mx = fmaxf(mx, v[j]);
  float s = 0.0f;
#pragma unroll
  for (int j = 0; j < 16; ++j) s += __expf(v[j] - mx);
  float lse = mx + __logf(s);
  float4* orow = (float4*)(out + (size_t)r * 16);
  orow[0] = make_float4(v[0] - lse, v[1] - lse, v[2] - lse, v[3] - lse);
  orow[1] = make_float4(v[4] - lse, v[5] - lse, v[6] - lse, v[7] - lse);
  orow[2] = make_float4(v[8] - lse, v[9] - lse, v[10] - lse, v[11] - lse);
  orow[3] = make_float4(v[12] - lse, v[13] - lse, v[14] - lse, v[15] - lse);
}

// ---------------- WMMA GEMM: Hout[n x HC] = X[n x 128] * Wsw(swizzled) ------
// blockDim.x = 32 * (HC/16). Each wave owns one 16-wide column tile and four
// 16-row subtiles (M=64/block); K=128 in four chunks -> 16 WMMA per wave.
template <int HC>
__global__ void k_gemm_wmma(const float* __restrict__ X, const _Float16* __restrict__ Wsw,
                            float* __restrict__ Hout, int n) {
  __shared__ _Float16 Alds[MROWS * 128];   // A tile, row-major [MROWS][128], f16

  const int tid = threadIdx.x;
  const int nth = blockDim.x;
  const int r0  = blockIdx.x * MROWS;

  // stage A tile: float4 global loads, f32 -> f16, one ds_store_b64 each
  for (int idx = tid; idx < MROWS * 32; idx += nth) {
    int r  = idx >> 5;
    int k4 = (idx & 31) << 2;
    float4 f = make_float4(0.f, 0.f, 0.f, 0.f);
    if (r0 + r < n) f = *(const float4*)(X + (size_t)(r0 + r) * 128 + k4);
    v4h hv;
    hv[0] = (_Float16)f.x; hv[1] = (_Float16)f.y;
    hv[2] = (_Float16)f.z; hv[3] = (_Float16)f.w;
    *(v4h*)(Alds + r * 128 + k4) = hv;
  }
  __syncthreads();

  const int wave = tid >> 5;
  const int lane = tid & 31;
  const int ct   = wave;          // column tile
  const int c0   = ct * 16;
  const int col  = lane & 15;     // A row within subtile / C column
  const int hi   = lane >> 4;     // lane half selector

  const v16h* __restrict__ Bf = (const v16h*)Wsw;

  v8f acc[4];
#pragma unroll
  for (int mt = 0; mt < 4; ++mt) acc[mt] = (v8f){0.f,0.f,0.f,0.f,0.f,0.f,0.f,0.f};

#pragma unroll
  for (int kc = 0; kc < 4; ++kc) {
    // B fragment: one contiguous 32B load (global_load_b128 x2), L1/L2 resident
    v16h b = Bf[(ct * 4 + kc) * 32 + lane];
#pragma unroll
    for (int mt = 0; mt < 4; ++mt) {
      // A fragment halves: K = kc*32+8*hi..+7 and K = kc*32+16+8*hi..+7
      const v8h* ap = (const v8h*)(Alds + (mt * 16 + col) * 128 + kc * 32 + 8 * hi);
      v8h a0 = ap[0];
      v8h a1 = ap[2];
      v16h a = __builtin_shufflevector(a0, a1, 0, 1, 2, 3, 4, 5, 6, 7,
                                                8, 9, 10, 11, 12, 13, 14, 15);
      acc[mt] = __builtin_amdgcn_wmma_f32_16x16x32_f16(false, a, false, b,
                                                       (short)0, acc[mt], false, false);
    }
  }

  // C/D layout: VGPR v -> M = v + 8*hi, N = lane&15
  if (r0 + MROWS <= n) {
    // full block: uniform branch, no per-store guards
    float* out0 = Hout + (size_t)(r0 + 8 * hi) * HC + c0 + col;
#pragma unroll
    for (int mt = 0; mt < 4; ++mt)
#pragma unroll
      for (int v = 0; v < 8; ++v)
        out0[(size_t)(mt * 16 + v) * HC] = acc[mt][v];
  } else {
#pragma unroll
    for (int mt = 0; mt < 4; ++mt)
#pragma unroll
      for (int v = 0; v < 8; ++v) {
        int r = r0 + mt * 16 + v + 8 * hi;
        if (r < n) Hout[(size_t)r * HC + c0 + col] = acc[mt][v];
      }
  }
}

// ---------------------------------------------------------------------------

extern "C" void kernel_launch(void* const* d_in, const int* in_sizes, int n_in,
                              void* d_out, int out_size, void* d_ws, size_t ws_size,
                              hipStream_t stream) {
  const float*     x   = (const float*)d_in[0];
  const long long* ei  = (const long long*)d_in[1];   // int64 per reference
  const float*     W1  = (const float*)d_in[2];
  const float*     b1  = (const float*)d_in[3];
  const float*     g1  = (const float*)d_in[4];
  const float*     be1 = (const float*)d_in[5];
  const float*     m1  = (const float*)d_in[6];
  const float*     v1  = (const float*)d_in[7];
  const float*     W2  = (const float*)d_in[8];
  const float*     b2  = (const float*)d_in[9];
  const float*     g2  = (const float*)d_in[10];
  const float*     be2 = (const float*)d_in[11];
  const float*     m2  = (const float*)d_in[12];
  const float*     v2  = (const float*)d_in[13];
  const float*     W3  = (const float*)d_in[14];
  const float*     b3  = (const float*)d_in[15];

  const int n = in_sizes[0] / 128;      // 50000
  const int E = in_sizes[1] / 2;        // 800000
  const long long* srcI = ei;           // edge_index[0]
  const long long* dstI = ei + E;       // edge_index[1]

  // workspace carve-out (64B-aligned sections)
  char* base = (char*)d_ws;
  size_t off = 0;
  auto take = [&](size_t bytes) -> char* {
    char* p = base + off;
    off = (off + bytes + 63) & ~(size_t)63;
    return p;
  };
  float* deg  = (float*)take((size_t)n * 4);
  float* dinv = (float*)take((size_t)n * 4);
  float* bufA = (float*)take((size_t)n * 128 * 4);
  float* bufB = (float*)take((size_t)n * 128 * 4);
  float* t16a = (float*)take((size_t)n * 16 * 4);
  float* t16b = (float*)take((size_t)n * 16 * 4);
  _Float16* W1s = (_Float16*)take(128 * 128 * 2);
  _Float16* W2s = (_Float16*)take(128 * 128 * 2);
  _Float16* W3s = (_Float16*)take(128 * 16 * 2);

  const int TB = 256;
  const int gN    = (n + TB - 1) / TB;
  const int gE    = (E + TB - 1) / TB;
  const int gNH4  = ((size_t)n * 32 + TB - 1) / TB;   // n*128/4 float4 elems
  const int gN4   = ((size_t)n * 4 + TB - 1) / TB;    // n*16/4  float4 elems
  const int gTile = (n + MROWS - 1) / MROWS;
  const int gScat = (E + 7) / 8;        // 8 waves (edges) per 256-thread block

  // degrees + normalization
  k_fill_deg<<<gN, TB, 0, stream>>>(deg, n);
  k_deg<<<gE, TB, 0, stream>>>(dstI, deg, E);
  k_dinv<<<gN, TB, 0, stream>>>(deg, dinv, n);

  // weights -> f16 fragment-swizzled
  k_swizzle_w<<<(128 * 128 + TB - 1) / TB, TB, 0, stream>>>(W1, W1s, 128);
  k_swizzle_w<<<(128 * 128 + TB - 1) / TB, TB, 0, stream>>>(W2, W2s, 128);
  k_swizzle_w<<<(128 * 16  + TB - 1) / TB, TB, 0, stream>>>(W3, W3s, 16);

  // ---- layer 1: h = x@W1 ; agg = self + edges + b1 ; relu(bn(agg)) ----
  k_gemm_wmma<128><<<gTile, 256, 0, stream>>>(x, W1s, bufA, n);
  k_selfinit<128><<<gNH4, TB, 0, stream>>>(bufA, dinv, b1, bufB, n);
  k_scatter<128><<<gScat, TB, 0, stream>>>(bufA, srcI, dstI, dinv, bufB, E);
  k_bn_relu<128><<<gNH4, TB, 0, stream>>>(bufB, g1, be1, m1, v1, bufA, n);

  // ---- layer 2 ----
  k_gemm_wmma<128><<<gTile, 256, 0, stream>>>(bufA, W2s, bufB, n);
  k_selfinit<128><<<gNH4, TB, 0, stream>>>(bufB, dinv, b2, bufA, n);
  k_scatter<128><<<gScat, TB, 0, stream>>>(bufB, srcI, dstI, dinv, bufA, E);
  k_bn_relu<128><<<gNH4, TB, 0, stream>>>(bufA, g2, be2, m2, v2, bufB, n);

  // ---- layer 3 (width 16) + log_softmax ----
  k_gemm_wmma<16><<<gTile, 32, 0, stream>>>(bufB, W3s, t16a, n);
  k_selfinit<16><<<gN4, TB, 0, stream>>>(t16a, dinv, b3, t16b, n);
  k_scatter<16><<<gScat, TB, 0, stream>>>(t16a, srcI, dstI, dinv, t16b, E);
  k_logsoftmax16<<<gN, TB, 0, stream>>>(t16b, (float*)d_out, n);
}
